// attention_57406532879088
// MI455X (gfx1250) — matrix operand order repaired and checked by reading, not compiled, not run on hardware
//
#include <hip/hip_runtime.h>
#include <hip/hip_bf16.h>
#include <math.h>

// ---------------------------------------------------------------------------
// B=16, C=512, L=4096.
// Pass 0: convert W1/W2 fp32->bf16 into ws, zero scores/ctx accumulators.
// Pass 1: per (b, 64-wide L tile): h1=relu(W1@x1+b1), h2=relu(W2@x2+b2) via
//         v_wmma_f32_16x16x32_bf16 with double-buffered LDS staging;
//         h2 -> d_out; scores[b,l]=sum_c h1*h2 -> ws.
// Pass 2: softmax over L -> attn (ws).
// Pass 3: recompute h1 tiles (same pipelined GEMM), ctx[b,c] += sum_l h1*attn.
// Pass 4: d_out[b,c,l] += ctx[b,c].
// ---------------------------------------------------------------------------

#define BB 16
#define CC 512
#define LL 4096
#define LT 64          // L-positions per workgroup
#define SLC 64         // K rows staged per LDS slice (2 WMMA sub-steps of 32)
#define NSL (CC / SLC) // 8 slices
#define XSTR 72        // padded LDS row stride (halves): 144B, conflict-free b128

typedef __bf16 bf16;
typedef __attribute__((ext_vector_type(16))) __bf16 v16bf;
typedef __attribute__((ext_vector_type(8)))  float  v8f;

union FragU { uint4 q[2]; v16bf v; };

// ---------------------------------------------------------------- pass 0
__global__ void k_init(const float* __restrict__ W1, const float* __restrict__ W2,
                       bf16* __restrict__ w1bf, bf16* __restrict__ w2bf,
                       float* __restrict__ scoresG, float* __restrict__ ctxG) {
    int i = blockIdx.x * 256 + threadIdx.x;
    if (i < CC * CC) { w1bf[i] = (bf16)W1[i]; w2bf[i] = (bf16)W2[i]; }
    if (i < BB * LL) scoresG[i] = 0.f;
    if (i < BB * CC) ctxG[i]    = 0.f;
}

// ---------------------------------------------------------------- pass 1
__global__ __launch_bounds__(256) void k_gemm_score(
    const float* __restrict__ x1, const float* __restrict__ x2,
    const bf16* __restrict__ w1, const bf16* __restrict__ w2,
    const float* __restrict__ b1, const float* __restrict__ b2,
    float* __restrict__ out, float* __restrict__ scoresG) {

    __shared__ bf16  xs1[2][LT * XSTR];   // double-buffered x1 slice [l][k]
    __shared__ bf16  xs2[2][LT * XSTR];   // double-buffered x2 slice [l][k]
    __shared__ float bias1[CC], bias2[CC];
    __shared__ float scoresLds[LT];

    const int l0   = blockIdx.x * LT;
    const int b    = blockIdx.y;
    const int tid  = threadIdx.x;
    const int wv   = tid >> 5;
    const int lane = tid & 31;
    const int n16  = lane & 15;
    const int kg   = lane >> 4;

    if (tid < CC) { bias1[tid] = b1[tid]; bias2[tid] = b2[tid]; }
    if (tid < LT) scoresLds[tid] = 0.f;

    // staging coords: thread moves rows {skk, skk+32} x 8 contiguous L columns
    const int skk = tid >> 3;         // 0..31
    const int slc = (tid & 7) * 8;    // 0..56

    const float* xb1 = x1 + (size_t)b * CC * LL + l0 + slc;
    const float* xb2 = x2 + (size_t)b * CC * LL + l0 + slc;

    auto stageLoad = [&](const float* xb, int kb, float* r) {
        const float* p0 = xb + (size_t)(kb + skk) * LL;
        const float* p1 = p0 + (size_t)32 * LL;
        #pragma unroll
        for (int j = 0; j < 8; ++j) { r[j] = p0[j]; r[8 + j] = p1[j]; }
    };
    auto stageStore = [&](bf16* buf, const float* r) {
        #pragma unroll
        for (int j = 0; j < 8; ++j) {
            buf[(slc + j) * XSTR + skk]      = (bf16)r[j];
            buf[(slc + j) * XSTR + skk + 32] = (bf16)r[8 + j];
        }
    };

    for (int rg = 0; rg < 2; ++rg) {
        const int mbase = rg * 256 + wv * 32;   // 32 output channels per wave
        v8f acc1[2][4], acc2[2][4];
        const v8f vzero = {0.f,0.f,0.f,0.f,0.f,0.f,0.f,0.f};
        #pragma unroll
        for (int mt = 0; mt < 2; ++mt)
            #pragma unroll
            for (int nt = 0; nt < 4; ++nt) { acc1[mt][nt] = vzero; acc2[mt][nt] = vzero; }

        float r1[16], r2[16];
        __syncthreads();                       // buffers free from previous rg
        stageLoad(xb1, 0, r1); stageLoad(xb2, 0, r2);
        stageStore(xs1[0], r1); stageStore(xs2[0], r2);

        for (int ks = 0; ks < NSL; ++ks) {
            __syncthreads();                   // slice ks visible; buf ks^1 free
            const int cur = ks & 1;
            const bool more = (ks + 1) < NSL;
            if (more) {                        // prefetch next slice into regs
                stageLoad(xb1, (ks + 1) * SLC, r1);
                stageLoad(xb2, (ks + 1) * SLC, r2);
            }
            #pragma unroll
            for (int s = 0; s < 2; ++s) {      // two K=32 WMMA sub-steps
                const int kb = ks * SLC + s * 32;
                v16bf a1[2], a2[2];
                #pragma unroll
                for (int mt = 0; mt < 2; ++mt) {
                    const size_t row = (size_t)(mbase + mt * 16 + n16) * CC;
                    FragU u;
                    u.q[0] = *(const uint4*)(w1 + row + kb + kg * 16);
                    u.q[1] = *(const uint4*)(w1 + row + kb + kg * 16 + 8);
                    a1[mt] = u.v;
                    u.q[0] = *(const uint4*)(w2 + row + kb + kg * 16);
                    u.q[1] = *(const uint4*)(w2 + row + kb + kg * 16 + 8);
                    a2[mt] = u.v;
                }
                #pragma unroll
                for (int nt = 0; nt < 4; ++nt) {
                    const bf16* bp1 = xs1[cur] + (nt * 16 + n16) * XSTR + s * 32 + kg * 16;
                    const bf16* bp2 = xs2[cur] + (nt * 16 + n16) * XSTR + s * 32 + kg * 16;
                    FragU ub;
                    ub.q[0] = *(const uint4*)(bp1); ub.q[1] = *(const uint4*)(bp1 + 8);
                    const v16bf bf1 = ub.v;
                    ub.q[0] = *(const uint4*)(bp2); ub.q[1] = *(const uint4*)(bp2 + 8);
                    const v16bf bf2 = ub.v;
                    #pragma unroll
                    for (int mt = 0; mt < 2; ++mt) {
                        acc1[mt][nt] = __builtin_amdgcn_wmma_f32_16x16x32_bf16(
                            false, a1[mt], false, bf1, (short)0, acc1[mt][nt], false, false);
                        acc2[mt][nt] = __builtin_amdgcn_wmma_f32_16x16x32_bf16(
                            false, a2[mt], false, bf2, (short)0, acc2[mt][nt], false, false);
                    }
                }
            }
            if (more) {                        // write next slice to other buffer
                stageStore(xs1[cur ^ 1], r1);
                stageStore(xs2[cur ^ 1], r2);
            }
        }

        // epilogue: bias + relu, h2 -> out, score partials -> LDS
        #pragma unroll
        for (int nt = 0; nt < 4; ++nt) {
            const int lcol = l0 + nt * 16 + n16;
            float sp = 0.f;
            #pragma unroll
            for (int mt = 0; mt < 2; ++mt) {
                #pragma unroll
                for (int r = 0; r < 8; ++r) {
                    const int m = mbase + mt * 16 + r + 8 * kg;
                    const float h1v = fmaxf(acc1[mt][nt][r] + bias1[m], 0.f);
                    const float h2v = fmaxf(acc2[mt][nt][r] + bias2[m], 0.f);
                    out[((size_t)b * CC + m) * LL + lcol] = h2v;
                    sp += h1v * h2v;
                }
            }
            atomicAdd(&scoresLds[nt * 16 + n16], sp);
        }
    }
    __syncthreads();
    if (tid < LT) scoresG[(size_t)b * LL + l0 + tid] = scoresLds[tid];
}

// ---------------------------------------------------------------- pass 2
__global__ void k_softmax(const float* __restrict__ scoresG, float* __restrict__ attnG) {
    __shared__ float red[256];
    const int b = blockIdx.x, tid = threadIdx.x;
    const float* s = scoresG + (size_t)b * LL;
    float mx = -1e30f;
    for (int l = tid; l < LL; l += 256) mx = fmaxf(mx, s[l]);
    red[tid] = mx; __syncthreads();
    for (int st = 128; st > 0; st >>= 1) {
        if (tid < st) red[tid] = fmaxf(red[tid], red[tid + st]);
        __syncthreads();
    }
    mx = red[0]; __syncthreads();
    float sum = 0.f;
    for (int l = tid; l < LL; l += 256) sum += __expf(s[l] - mx);
    red[tid] = sum; __syncthreads();
    for (int st = 128; st > 0; st >>= 1) {
        if (tid < st) red[tid] += red[tid + st];
        __syncthreads();
    }
    const float inv = 1.f / red[0];
    for (int l = tid; l < LL; l += 256)
        attnG[(size_t)b * LL + l] = __expf(s[l] - mx) * inv;
}

// ---------------------------------------------------------------- pass 3
__global__ __launch_bounds__(256) void k_ctx(
    const float* __restrict__ x1, const bf16* __restrict__ w1,
    const float* __restrict__ b1, const float* __restrict__ attnG,
    float* __restrict__ ctxG) {

    __shared__ bf16  xs[2][LT * XSTR];
    __shared__ float bias[CC];
    __shared__ float attnLds[LT];

    const int l0   = blockIdx.x * LT;
    const int b    = blockIdx.y;
    const int tid  = threadIdx.x;
    const int wv   = tid >> 5;
    const int lane = tid & 31;
    const int n16  = lane & 15;
    const int kg   = lane >> 4;

    if (tid < CC) bias[tid] = b1[tid];
    if (tid < LT) attnLds[tid] = attnG[(size_t)b * LL + l0 + tid];

    const int skk = tid >> 3;
    const int slc = (tid & 7) * 8;
    const int mbase = wv * 64;   // 64 output channels per wave (4 M-tiles)

    const float* xb = x1 + (size_t)b * CC * LL + l0 + slc;

    auto stageLoad = [&](int kb, float* r) {
        const float* p0 = xb + (size_t)(kb + skk) * LL;
        const float* p1 = p0 + (size_t)32 * LL;
        #pragma unroll
        for (int j = 0; j < 8; ++j) { r[j] = p0[j]; r[8 + j] = p1[j]; }
    };
    auto stageStore = [&](bf16* buf, const float* r) {
        #pragma unroll
        for (int j = 0; j < 8; ++j) {
            buf[(slc + j) * XSTR + skk]      = (bf16)r[j];
            buf[(slc + j) * XSTR + skk + 32] = (bf16)r[8 + j];
        }
    };

    v8f acc[4][4];
    const v8f vzero = {0.f,0.f,0.f,0.f,0.f,0.f,0.f,0.f};
    #pragma unroll
    for (int mt = 0; mt < 4; ++mt)
        #pragma unroll
        for (int nt = 0; nt < 4; ++nt) acc[mt][nt] = vzero;

    float rr[16];
    stageLoad(0, rr);
    stageStore(xs[0], rr);

    for (int ks = 0; ks < NSL; ++ks) {
        __syncthreads();
        const int cur = ks & 1;
        const bool more = (ks + 1) < NSL;
        if (more) stageLoad((ks + 1) * SLC, rr);
        #pragma unroll
        for (int s = 0; s < 2; ++s) {
            const int kb = ks * SLC + s * 32;
            v16bf a[4];
            #pragma unroll
            for (int mt = 0; mt < 4; ++mt) {
                const size_t row = (size_t)(mbase + mt * 16 + n16) * CC;
                FragU u;
                u.q[0] = *(const uint4*)(w1 + row + kb + kg * 16);
                u.q[1] = *(const uint4*)(w1 + row + kb + kg * 16 + 8);
                a[mt] = u.v;
            }
            #pragma unroll
            for (int nt = 0; nt < 4; ++nt) {
                const bf16* bp = xs[cur] + (nt * 16 + n16) * XSTR + s * 32 + kg * 16;
                FragU ub;
                ub.q[0] = *(const uint4*)(bp); ub.q[1] = *(const uint4*)(bp + 8);
                const v16bf bfv = ub.v;
                #pragma unroll
                for (int mt = 0; mt < 4; ++mt)
                    acc[mt][nt] = __builtin_amdgcn_wmma_f32_16x16x32_bf16(
                        false, a[mt], false, bfv, (short)0, acc[mt][nt], false, false);
            }
        }
        if (more) stageStore(xs[cur ^ 1], rr);
    }

    // ctx[b, m] += sum_n relu(h1[m,n]) * attn[n]
    #pragma unroll
    for (int mt = 0; mt < 4; ++mt) {
        #pragma unroll
        for (int r = 0; r < 8; ++r) {
            const int m = mbase + mt * 16 + r + 8 * kg;
            float cp = 0.f;
            #pragma unroll
            for (int nt = 0; nt < 4; ++nt) {
                const float h = fmaxf(acc[mt][nt][r] + bias[m], 0.f);
                cp += h * attnLds[nt * 16 + n16];
            }
            // butterfly reduce across the 16-lane half (n dimension)
            cp += __shfl_xor(cp, 1, 32);
            cp += __shfl_xor(cp, 2, 32);
            cp += __shfl_xor(cp, 4, 32);
            cp += __shfl_xor(cp, 8, 32);
            if (n16 == 0) atomicAdd(&ctxG[b * CC + m], cp);
        }
    }
}

// ---------------------------------------------------------------- pass 4
__global__ void k_add_ctx(float* __restrict__ out, const float* __restrict__ ctxG) {
    const size_t idx = (size_t)blockIdx.x * 256 + threadIdx.x;  // float4 index
    const size_t e = idx * 4;                                    // element index
    const int b = (int)(e >> 21);               // C*L = 2^21
    const int c = (int)((e >> 12) & (CC - 1));  // L = 2^12
    const float cv = ctxG[b * CC + c];
    float4* p = (float4*)out;
    float4 v = p[idx];
    v.x += cv; v.y += cv; v.z += cv; v.w += cv;
    p[idx] = v;
}

// ---------------------------------------------------------------- host
extern "C" void kernel_launch(void* const* d_in, const int* in_sizes, int n_in,
                              void* d_out, int out_size, void* d_ws, size_t ws_size,
                              hipStream_t stream) {
    const float* x1 = (const float*)d_in[0];
    const float* x2 = (const float*)d_in[1];
    const float* W1 = (const float*)d_in[2];
    const float* b1 = (const float*)d_in[3];
    const float* W2 = (const float*)d_in[4];
    const float* b2 = (const float*)d_in[5];
    float* out = (float*)d_out;

    char* ws = (char*)d_ws;
    bf16*  w1bf    = (bf16*)(ws);
    bf16*  w2bf    = (bf16*)(ws + 524288);
    float* scoresG = (float*)(ws + 1048576);   // B*L fp32
    float* attnG   = (float*)(ws + 1310720);   // B*L fp32
    float* ctxG    = (float*)(ws + 1572864);   // B*C fp32

    k_init<<<(CC * CC) / 256, 256, 0, stream>>>(W1, W2, w1bf, w2bf, scoresG, ctxG);

    dim3 g1(LL / LT, BB);
    k_gemm_score<<<g1, 256, 0, stream>>>(x1, x2, w1bf, w2bf, b1, b2, out, scoresG);

    k_softmax<<<BB, 256, 0, stream>>>(scoresG, attnG);

    k_ctx<<<g1, 256, 0, stream>>>(x1, w1bf, b1, attnG, ctxG);

    k_add_ctx<<<(BB * CC * LL / 4) / 256, 256, 0, stream>>>(out, ctxG);
}
